// ConstellationNet_85796266705422
// MI455X (gfx1250) — matrix operand order repaired and checked by reading, not compiled, not run on hardware
//
#include <hip/hip_runtime.h>

// Problem constants (from setup_inputs: B=64, H=40, C=128, K=128, epoch=10)
#define C_DIM   128
#define K_CL    128
#define N_ROWS  (64 * 40 * 40)   // 102400 flattened cells
#define BETA    100.0f
#define LBDA    1.0f
#define EPOCHS  10

#define ROWS_PB   32             // rows of x per block (2 row-tiles of 16)
#define XT_STRIDE 136            // f16 tile row stride (pad: 16 A-loads hit distinct banks)
#define DT_STRIDE 132            // f32 d-tile row stride

typedef __attribute__((ext_vector_type(16))) _Float16 v16h;
typedef __attribute__((ext_vector_type(8)))  float    v8f;

union AFrag { v16h v; uint4 q[2]; };

// ---------------------------------------------------------------------------
// init: x_head_sum[k] = sum_c x[k,c]; v = v_init (fp32 + f16 copy); v2 = ||v||^2
// s = 0; msum = 0.   One block of 128 threads (thread k owns cluster k).
// ---------------------------------------------------------------------------
__global__ void ck_init(const float* __restrict__ x, const float* __restrict__ vinit,
                        float* __restrict__ v, _Float16* __restrict__ vh,
                        float* __restrict__ v2, float* __restrict__ s,
                        float* __restrict__ msum, float* __restrict__ xhead) {
    const int k = threadIdx.x;
    float hs = 0.f, nrm = 0.f;
    for (int c = 0; c < C_DIM; ++c) {
        hs += x[k * C_DIM + c];
        float vv = vinit[k * C_DIM + c];
        v[k * C_DIM + c]  = vv;
        vh[k * C_DIM + c] = (_Float16)vv;
        nrm += vv * vv;
    }
    xhead[k] = hs;
    v2[k]    = nrm;
    s[k]     = 0.f;
    msum[k]  = 0.f;
}

// ---------------------------------------------------------------------------
// one-time x (fp32) -> x_h (f16) conversion.  8 elements / thread.
// ---------------------------------------------------------------------------
__global__ void ck_convx(const float* __restrict__ x, _Float16* __restrict__ xh) {
    const long i = ((long)blockIdx.x * 256 + threadIdx.x) * 8;
    float4 a = *(const float4*)(x + i);
    float4 b = *(const float4*)(x + i + 4);
    _Float16 h[8] = {(_Float16)a.x, (_Float16)a.y, (_Float16)a.z, (_Float16)a.w,
                     (_Float16)b.x, (_Float16)b.y, (_Float16)b.z, (_Float16)b.w};
    *(uint4*)(xh + i) = *(const uint4*)h;
}

// ---------------------------------------------------------------------------
// main iteration kernel.  Block = 32 rows of x, all 128 clusters, 8 waves.
// Wave w owns clusters [16w,16w+16): B fragments preloaded once (reused for
// both 16-row tiles); per tile 4x wmma_f32_16x16x32_f16 with all fragments
// preloaded (single wait, back-to-back WMMAs).  Then d = sqrt(max(.,0)),
// row softmax(-beta*d), column sums -> atomic msum.  Last epoch writes d.
// ---------------------------------------------------------------------------
__launch_bounds__(256, 2)
__global__ void ck_main(const void* __restrict__ xsrc, int x_is_f16,
                        const _Float16* __restrict__ vh,
                        const float* __restrict__ v2,
                        float* __restrict__ msum,
                        float* __restrict__ dout, int write_d) {
    __shared__ __align__(16) _Float16 xt[ROWS_PB * XT_STRIDE];
    __shared__ __align__(16) float    dt[ROWS_PB * DT_STRIDE];
    __shared__ float x2s[ROWS_PB];

    const int  t    = threadIdx.x;
    const int  lane = t & 31;
    const int  wave = t >> 5;
    const long rowBase = (long)blockIdx.x * ROWS_PB;

    // ---- stage x tile (32 x 128) into LDS as f16, fused ||x||^2 ----
    {
        const int r   = t >> 3;          // 0..31
        const int c16 = (t & 7) * 16;    // 0,16,...,112
        float xv[16];
        if (x_is_f16) {
            const _Float16* src = (const _Float16*)xsrc + (rowBase + r) * C_DIM + c16;
            uint4 q0 = *(const uint4*)(src);
            uint4 q1 = *(const uint4*)(src + 8);
            *(uint4*)(&xt[r * XT_STRIDE + c16])     = q0;
            *(uint4*)(&xt[r * XT_STRIDE + c16 + 8]) = q1;
            const _Float16* h = (const _Float16*)&q0;
            #pragma unroll
            for (int j = 0; j < 8; ++j) xv[j] = (float)h[j];
            h = (const _Float16*)&q1;
            #pragma unroll
            for (int j = 0; j < 8; ++j) xv[8 + j] = (float)h[j];
        } else {
            const float* src = (const float*)xsrc + (rowBase + r) * C_DIM + c16;
            #pragma unroll
            for (int q = 0; q < 4; ++q) {
                float4 a = *(const float4*)(src + q * 4);
                xv[q * 4 + 0] = a.x; xv[q * 4 + 1] = a.y;
                xv[q * 4 + 2] = a.z; xv[q * 4 + 3] = a.w;
            }
            _Float16 h[16];
            #pragma unroll
            for (int j = 0; j < 16; ++j) h[j] = (_Float16)xv[j];
            *(uint4*)(&xt[r * XT_STRIDE + c16])     = ((const uint4*)h)[0];
            *(uint4*)(&xt[r * XT_STRIDE + c16 + 8]) = ((const uint4*)h)[1];
        }
        float ss = 0.f;
        #pragma unroll
        for (int j = 0; j < 16; ++j) ss += xv[j] * xv[j];
        // reduce across the 8 lanes sharing this row (lane bits [2:0])
        ss += __shfl_xor(ss, 1, 32);
        ss += __shfl_xor(ss, 2, 32);
        ss += __shfl_xor(ss, 4, 32);
        if ((t & 7) == 0) x2s[r] = ss;
    }
    __syncthreads();

    // ---- WMMA ----
    const int colBase = wave * 16;
    const int hi = lane >> 4;   // lane half
    const int lo = lane & 15;

    // B fragments (cluster slab), one contiguous 32B load / lane / kstep:
    // lanes<16 hold K c0..c0+15, lanes>=16 hold K c0+16..c0+31 for col N=lo.
    v16h bf[4];
    #pragma unroll
    for (int ks = 0; ks < 4; ++ks)
        bf[ks] = *(const v16h*)(vh + (colBase + lo) * C_DIM + ks * 32 + hi * 16);

    const float v2n = v2[colBase + lo];

    #pragma unroll
    for (int rt = 0; rt < 2; ++rt) {
        const int rowOff = rt * 16;
        // A fragments: row M=lo; lanes<16 K c0..c0+15,
        //              lanes>=16 K c0+16..c0+31
        AFrag af[4];
        #pragma unroll
        for (int ks = 0; ks < 4; ++ks) {
            const _Float16* ap = &xt[(rowOff + lo) * XT_STRIDE + ks * 32 + hi * 16];
            af[ks].q[0] = *(const uint4*)(ap);
            af[ks].q[1] = *(const uint4*)(ap + 8);
        }
        v8f acc = {};
        #pragma unroll
        for (int ks = 0; ks < 4; ++ks)
            acc = __builtin_amdgcn_wmma_f32_16x16x32_f16(
                      false, af[ks].v, false, bf[ks],
                      (short)0, acc, false, false);

        // C/D layout: lane holds N=lo, rows r + 8*hi
        #pragma unroll
        for (int r = 0; r < 8; ++r) {
            const int m = rowOff + r + hi * 8;
            float d2 = x2s[m] + v2n - 2.0f * acc[r];
            float dv = __builtin_amdgcn_sqrtf(fmaxf(d2, 0.0f));
            dt[m * DT_STRIDE + colBase + lo] = dv;
            if (write_d)
                dout[(rowBase + m) * K_CL + colBase + lo] = dv;
        }
    }
    __syncthreads();

    // ---- row softmax of -beta*d (wave w: rows 4w..4w+3; 4 cols/lane) ----
    #pragma unroll
    for (int rr = 0; rr < 4; ++rr) {
        const int r = wave * 4 + rr;
        float* row = &dt[r * DT_STRIDE];
        float vbuf[4], dmin = 1e30f;
        #pragma unroll
        for (int j = 0; j < 4; ++j) { vbuf[j] = row[lane * 4 + j]; dmin = fminf(dmin, vbuf[j]); }
        for (int off = 16; off; off >>= 1) dmin = fminf(dmin, __shfl_xor(dmin, off, 32));
        float e[4], ssum = 0.f;
        #pragma unroll
        for (int j = 0; j < 4; ++j) { e[j] = __expf(BETA * (dmin - vbuf[j])); ssum += e[j]; }
        for (int off = 16; off; off >>= 1) ssum += __shfl_xor(ssum, off, 32);
        const float inv = 1.0f / ssum;
        #pragma unroll
        for (int j = 0; j < 4; ++j) row[lane * 4 + j] = e[j] * inv;
    }
    __syncthreads();

    // ---- column sums of m over this 32-row tile -> global msum ----
    if (t < K_CL) {
        float cs = 0.f;
        #pragma unroll
        for (int m = 0; m < ROWS_PB; ++m) cs += dt[m * DT_STRIDE + t];
        atomicAdd(&msum[t], cs);
    }
}

// ---------------------------------------------------------------------------
// v update: mu = lbda/(s+msum); v_p = x_head_sum broadcast;
// v = (1-mu)*v + mu*v_p; refresh v_h and ||v||^2; s += msum; msum = 0.
// ---------------------------------------------------------------------------
__global__ void ck_update(float* __restrict__ v, _Float16* __restrict__ vh,
                          float* __restrict__ v2, float* __restrict__ s,
                          float* __restrict__ msum, const float* __restrict__ xhead) {
    const int k  = threadIdx.x;
    const float ms = msum[k];
    const float mu = LBDA / (s[k] + ms);
    const float vp = xhead[k];
    float nrm = 0.f;
    for (int c = 0; c < C_DIM; ++c) {
        float nv = (1.0f - mu) * v[k * C_DIM + c] + mu * vp;
        v[k * C_DIM + c]  = nv;
        vh[k * C_DIM + c] = (_Float16)nv;
        nrm += nv * nv;
    }
    v2[k]   = nrm;
    s[k]   += ms;
    msum[k] = 0.f;
}

// ---------------------------------------------------------------------------
extern "C" void kernel_launch(void* const* d_in, const int* in_sizes, int n_in,
                              void* d_out, int out_size, void* d_ws, size_t ws_size,
                              hipStream_t stream) {
    const float* x     = (const float*)d_in[0];   // [N, C] fp32
    const float* vinit = (const float*)d_in[1];   // [K, C] fp32
    float*       out   = (float*)d_out;           // [N, K] fp32 (last-iteration d)

    // workspace layout (all offsets 512B-aligned)
    char* ws = (char*)d_ws;
    float*    v     = (float*)(ws);                                // 64 KB
    float*    v2    = (float*)(ws + 65536);                        // 512 B
    float*    s     = (float*)(ws + 65536 + 512);                  // 512 B
    float*    msum  = (float*)(ws + 65536 + 1024);                 // 512 B
    float*    xhead = (float*)(ws + 65536 + 1536);                 // 512 B
    _Float16* vh    = (_Float16*)(ws + 65536 + 2048);              // 32 KB
    const size_t baseBytes = 65536 + 2048 + 32768;                 // 100352
    _Float16* xh    = (_Float16*)(ws + baseBytes);                 // N*C*2 = 25 MiB
    const size_t needXh = baseBytes + (size_t)N_ROWS * C_DIM * sizeof(_Float16);
    const int useXh = (ws_size >= needXh) ? 1 : 0;

    ck_init<<<1, 128, 0, stream>>>(x, vinit, v, vh, v2, s, msum, xhead);
    if (useXh)
        ck_convx<<<(N_ROWS * C_DIM) / 2048, 256, 0, stream>>>(x, xh);

    const void* xsrc = useXh ? (const void*)xh : (const void*)x;
    for (int e = 0; e < EPOCHS; ++e) {
        ck_main<<<N_ROWS / ROWS_PB, 256, 0, stream>>>(xsrc, useXh, vh, v2, msum,
                                                      out, (e == EPOCHS - 1) ? 1 : 0);
        if (e != EPOCHS - 1)
            ck_update<<<1, 128, 0, stream>>>(v, vh, v2, s, msum, xhead);
    }
}
